// APPNP_JJ_21474836480044
// MI455X (gfx1250) — matrix operand order repaired and checked by reading, not compiled, not run on hardware
//
#include <hip/hip_runtime.h>
#include <hip/hip_fp16.h>

typedef __attribute__((ext_vector_type(16))) _Float16 v16h;
typedef __attribute__((ext_vector_type(8)))  float    v8f;

#define N_NODES  50000
#define N_EDGES  625000
#define IN_SIZE  128
#define HID_SIZE 128
#define OUT_SIZE 40
#define OUT_PAD  48
#define NUM_TIME 40
#define K_HOPS   2
#define ALPHA    0.5f
#define EPS      1e-5f

// ---------------------------------------------------------------------------
// Utility
// ---------------------------------------------------------------------------
__global__ void fill_zero_f32(float* __restrict__ p, int n) {
    int i = blockIdx.x * blockDim.x + threadIdx.x;
    if (i < n) p[i] = 0.0f;
}

// ---------------------------------------------------------------------------
// Degrees + norms
// ---------------------------------------------------------------------------
__global__ void degrees_kernel(const int* __restrict__ src, const int* __restrict__ dst,
                               float* __restrict__ deg_out, float* __restrict__ deg_in) {
    int e = blockIdx.x * blockDim.x + threadIdx.x;
    if (e < N_EDGES) {
        atomicAdd(&deg_out[src[e]], 1.0f);
        atomicAdd(&deg_in[dst[e]], 1.0f);
    }
}

__global__ void norms_kernel(float* __restrict__ deg_out, float* __restrict__ deg_in) {
    int i = blockIdx.x * blockDim.x + threadIdx.x;
    if (i < N_NODES) {
        deg_out[i] = rsqrtf(fmaxf(deg_out[i], 1.0f));
        deg_in[i]  = rsqrtf(fmaxf(deg_in[i], 1.0f));
    }
}

// ---------------------------------------------------------------------------
// Propagation: gather h[src]*norm_out[src], scatter-add into agg[dst]
// 32 lanes per edge, float4 per lane (128 feats)
// ---------------------------------------------------------------------------
__global__ void scatter_kernel(const float* __restrict__ h,
                               const int* __restrict__ src, const int* __restrict__ dst,
                               const float* __restrict__ norm_out,
                               float* __restrict__ agg) {
    int t = blockIdx.x * blockDim.x + threadIdx.x;
    int e = t >> 5;
    int lane = t & 31;
    if (e < N_EDGES) {
        int s = src[e];
        int d = dst[e];
        float no = norm_out[s];
        const float4* hp = (const float4*)(h + (size_t)s * IN_SIZE);
        float4 v = hp[lane];
        float* ap = agg + (size_t)d * IN_SIZE + lane * 4;
        atomicAdd(ap + 0, v.x * no);
        atomicAdd(ap + 1, v.y * no);
        atomicAdd(ap + 2, v.z * no);
        atomicAdd(ap + 3, v.w * no);
    }
}

__global__ void update_kernel(const float* __restrict__ agg, const float* __restrict__ h0,
                              const float* __restrict__ norm_in, float* __restrict__ h) {
    int i = blockIdx.x * blockDim.x + threadIdx.x;
    if (i < N_NODES * IN_SIZE) {
        int node = i >> 7;
        h[i] = (1.0f - ALPHA) * agg[i] * norm_in[node] + ALPHA * h0[i];
    }
}

// ---------------------------------------------------------------------------
// Per-time-group standardization
// ---------------------------------------------------------------------------
__global__ void time_stats_kernel(const float* __restrict__ h, const int* __restrict__ times,
                                  float* __restrict__ sum, float* __restrict__ sumsq,
                                  float* __restrict__ cnt) {
    int i = blockIdx.x * blockDim.x + threadIdx.x;
    if (i < N_NODES * IN_SIZE) {
        int node = i >> 7;
        int d = i & 127;
        int t = times[node];
        float v = h[i];
        atomicAdd(&sum[t * IN_SIZE + d], v);
        atomicAdd(&sumsq[t * IN_SIZE + d], v * v);
        if (d == 0) atomicAdd(&cnt[t], 1.0f);
    }
}

__global__ void time_norm_kernel(const float* __restrict__ h, const int* __restrict__ times,
                                 const float* __restrict__ sum, const float* __restrict__ sumsq,
                                 const float* __restrict__ cnt, _Float16* __restrict__ h16) {
    int i = blockIdx.x * blockDim.x + threadIdx.x;
    if (i < N_NODES * IN_SIZE) {
        int node = i >> 7;
        int d = i & 127;
        int t = times[node];
        float c = fmaxf(cnt[t], 1.0f);
        float mean = sum[t * IN_SIZE + d] / c;
        float var = sumsq[t * IN_SIZE + d] / c - mean * mean;
        float x = (h[i] - mean) * rsqrtf(fmaxf(var, 0.0f) + EPS);
        h16[i] = (_Float16)x;
    }
}

// ---------------------------------------------------------------------------
// Weight conversion f32 -> f16 (w2 padded to 48 rows)
// ---------------------------------------------------------------------------
__global__ void cvt_w1_kernel(const float* __restrict__ w1, _Float16* __restrict__ w1h) {
    int i = blockIdx.x * blockDim.x + threadIdx.x;
    if (i < HID_SIZE * IN_SIZE) w1h[i] = (_Float16)w1[i];
}

__global__ void cvt_w2_kernel(const float* __restrict__ w2, _Float16* __restrict__ w2h) {
    int i = blockIdx.x * blockDim.x + threadIdx.x;
    if (i < OUT_PAD * HID_SIZE) {
        int n = i >> 7;
        int k = i & 127;
        w2h[i] = (n < OUT_SIZE) ? (_Float16)w2[n * HID_SIZE + k] : (_Float16)0.0f;
    }
}

// ---------------------------------------------------------------------------
// WMMA 16x16 output tile, K-accumulation over Ktot (f16 in, f32 accum).
// A: row-major [*, Ktot] activations. B: row-major [n][k] weights (B^T layout,
// i.e. B matrix element (K,N) = W[n][k]).
//
// A per-lane layout (ISA 16-bit A 16x32): lane l, row = m0 + (l&15);
// VGPR v (pair of halfs) holds K = kb + (v>=4?16:0) + (l>>4)*8 + (v&3)*2.
// B per-lane layout: lane l, col n = n0 + (l&15); same K order as A:
// VGPR v holds K = kb + (v>=4?16:0) + (l>>4)*8 + (v&3)*2.
// C/D: lane l col n0+(l&15); element r holds row m0 + r + (l>>4)*8.
// ---------------------------------------------------------------------------
__device__ __forceinline__ v8f wmma_gemm_tile(const _Float16* __restrict__ A, int m0,
                                              const _Float16* __restrict__ B, int n0,
                                              int Ktot) {
    v8f acc = {};
    int lane = threadIdx.x & 31;
    int hsel = lane >> 4;               // 0 or 1
    int row  = m0 + (lane & 15);
    int nn   = n0 + (lane & 15);
    const unsigned* arow = (const unsigned*)(A + (size_t)row * Ktot);
    for (int kb = 0; kb < Ktot; kb += 32) {
        union { v16h v; unsigned u[8]; } a, b;
#pragma unroll
        for (int v = 0; v < 8; ++v) {
            int kp = kb + ((v >= 4) ? 16 : 0) + hsel * 8 + (v & 3) * 2;
            a.u[v] = arow[kp >> 1];
        }
        const unsigned* brow = (const unsigned*)(B + (size_t)nn * Ktot);
#pragma unroll
        for (int v = 0; v < 8; ++v) b.u[v] = brow[(kb + ((v >= 4) ? 16 : 0) + hsel * 8 + (v & 3) * 2) >> 1];
        acc = __builtin_amdgcn_wmma_f32_16x16x32_f16(
            /*neg_a=*/false, a.v, /*neg_b=*/false, b.v,
            /*c_mod=*/(short)0, acc, /*reuse_a=*/false, /*reuse_b=*/false);
    }
    return acc;
}

// Layer 1: hid16[N,128] = relu(h16 @ w1^T + b1), f16 out for next GEMM.
// One wave per 16x16 tile; 3125 M-tiles x 8 N-tiles = 25000 waves.
__global__ void mlp1_kernel(const _Float16* __restrict__ h16, const _Float16* __restrict__ w1h,
                            const float* __restrict__ b1, _Float16* __restrict__ hid16) {
    int wave = (blockIdx.x * blockDim.x + threadIdx.x) >> 5;
    int lane = threadIdx.x & 31;
    int mtile = wave >> 3;
    int ntile = wave & 7;
    if (mtile >= N_NODES / 16) return;  // wave-uniform
    int m0 = mtile * 16, n0 = ntile * 16;
    v8f acc = wmma_gemm_tile(h16, m0, w1h, n0, IN_SIZE);
    int nn = n0 + (lane & 15);
    float bias = b1[nn];
    int mbase = m0 + (lane >> 4) * 8;
#pragma unroll
    for (int r = 0; r < 8; ++r) {
        float v = acc[r] + bias;
        v = fmaxf(v, 0.0f);
        hid16[(size_t)(mbase + r) * HID_SIZE + nn] = (_Float16)v;
    }
}

// Layer 2: out[N,40] = sigmoid(hid16 @ w2^T + b2). N-tiles padded to 48.
__global__ void mlp2_kernel(const _Float16* __restrict__ hid16, const _Float16* __restrict__ w2h,
                            const float* __restrict__ b2, float* __restrict__ out) {
    int wave = (blockIdx.x * blockDim.x + threadIdx.x) >> 5;
    int lane = threadIdx.x & 31;
    int mtile = wave / 3;
    int ntile = wave % 3;
    if (mtile >= N_NODES / 16) return;  // wave-uniform
    int m0 = mtile * 16, n0 = ntile * 16;
    v8f acc = wmma_gemm_tile(hid16, m0, w2h, n0, HID_SIZE);
    int nn = n0 + (lane & 15);
    if (nn < OUT_SIZE) {
        float bias = b2[nn];
        int mbase = m0 + (lane >> 4) * 8;
#pragma unroll
        for (int r = 0; r < 8; ++r) {
            float v = acc[r] + bias;
            v = 1.0f / (1.0f + expf(-v));
            out[(size_t)(mbase + r) * OUT_SIZE + nn] = v;
        }
    }
}

// ---------------------------------------------------------------------------
// Host launcher
// ---------------------------------------------------------------------------
static inline int cdiv(long long a, long long b) { return (int)((a + b - 1) / b); }

extern "C" void kernel_launch(void* const* d_in, const int* in_sizes, int n_in,
                              void* d_out, int out_size, void* d_ws, size_t ws_size,
                              hipStream_t stream) {
    const float* features = (const float*)d_in[0];
    const float* w1       = (const float*)d_in[1];
    const float* b1       = (const float*)d_in[2];
    const float* w2       = (const float*)d_in[3];
    const float* b2       = (const float*)d_in[4];
    const int*   src      = (const int*)d_in[5];
    const int*   dst      = (const int*)d_in[6];
    const int*   times    = (const int*)d_in[7];
    float* out = (float*)d_out;

    // Workspace carve-out (256-byte aligned)
    char* ws = (char*)d_ws;
    size_t off = 0;
    auto alloc = [&](size_t bytes) -> void* {
        void* p = ws + off;
        off = (off + bytes + 255) & ~(size_t)255;
        return p;
    };
    float*    deg_out = (float*)alloc((size_t)N_NODES * 4);
    float*    deg_in  = (float*)alloc((size_t)N_NODES * 4);
    float*    h       = (float*)alloc((size_t)N_NODES * IN_SIZE * 4);
    float*    agg     = (float*)alloc((size_t)N_NODES * IN_SIZE * 4);
    float*    sum     = (float*)alloc((size_t)NUM_TIME * IN_SIZE * 4);
    float*    sumsq   = (float*)alloc((size_t)NUM_TIME * IN_SIZE * 4);
    float*    cnt     = (float*)alloc((size_t)NUM_TIME * 4);
    _Float16* h16     = (_Float16*)alloc((size_t)N_NODES * IN_SIZE * 2);
    _Float16* hid16   = (_Float16*)alloc((size_t)N_NODES * HID_SIZE * 2);
    _Float16* w1h     = (_Float16*)alloc((size_t)HID_SIZE * IN_SIZE * 2);
    _Float16* w2h     = (_Float16*)alloc((size_t)OUT_PAD * HID_SIZE * 2);
    (void)ws_size; (void)in_sizes; (void)n_in; (void)out_size;

    const int B = 256;

    // 1) degrees -> norms
    fill_zero_f32<<<cdiv(N_NODES, B), B, 0, stream>>>(deg_out, N_NODES);
    fill_zero_f32<<<cdiv(N_NODES, B), B, 0, stream>>>(deg_in, N_NODES);
    degrees_kernel<<<cdiv(N_EDGES, B), B, 0, stream>>>(src, dst, deg_out, deg_in);
    norms_kernel<<<cdiv(N_NODES, B), B, 0, stream>>>(deg_out, deg_in);

    // 2) K_HOPS propagation
    const long long NF = (long long)N_NODES * IN_SIZE;
    for (int k = 0; k < K_HOPS; ++k) {
        const float* h_cur = (k == 0) ? features : h;
        fill_zero_f32<<<cdiv(NF, B), B, 0, stream>>>(agg, (int)NF);
        scatter_kernel<<<cdiv((long long)N_EDGES * 32, B), B, 0, stream>>>(
            h_cur, src, dst, deg_out /*norm_out*/, agg);
        update_kernel<<<cdiv(NF, B), B, 0, stream>>>(agg, features, deg_in /*norm_in*/, h);
    }

    // 3) per-time standardization -> f16
    fill_zero_f32<<<cdiv(NUM_TIME * IN_SIZE, B), B, 0, stream>>>(sum, NUM_TIME * IN_SIZE);
    fill_zero_f32<<<cdiv(NUM_TIME * IN_SIZE, B), B, 0, stream>>>(sumsq, NUM_TIME * IN_SIZE);
    fill_zero_f32<<<1, B, 0, stream>>>(cnt, NUM_TIME);
    time_stats_kernel<<<cdiv(NF, B), B, 0, stream>>>(h, times, sum, sumsq, cnt);
    time_norm_kernel<<<cdiv(NF, B), B, 0, stream>>>(h, times, sum, sumsq, cnt, h16);

    // 4) weights -> f16
    cvt_w1_kernel<<<cdiv(HID_SIZE * IN_SIZE, B), B, 0, stream>>>(w1, w1h);
    cvt_w2_kernel<<<cdiv(OUT_PAD * HID_SIZE, B), B, 0, stream>>>(w2, w2h);

    // 5) MLP via WMMA
    {
        long long waves1 = (long long)(N_NODES / 16) * 8;       // 25000
        mlp1_kernel<<<cdiv(waves1 * 32, B), B, 0, stream>>>(h16, w1h, b1, hid16);
        long long waves2 = (long long)(N_NODES / 16) * 3;       // 9375
        mlp2_kernel<<<cdiv(waves2 * 32, B), B, 0, stream>>>(hid16, w2h, b2, out);
    }
}
